// AttentionConv_60730837566325
// MI455X (gfx1250) — compile-verified, matrix operand order checked by reading, not run on hardware
//
#include <hip/hip_runtime.h>
#include <hip/hip_bf16.h>

// ---------------------------------------------------------------------------
// AttentionConv for MI455X (gfx1250, wave32, WMMA + async-to-LDS pipeline).
//   Stage 0: pack x / {wq,wk,wv} to bf16 in WMMA per-lane operand layout
//            (each lane's 16 bf16 operand = 32 contiguous bytes).
//   Stage 1: fused q/k/v GEMMs: double-buffered GLOBAL_LOAD_ASYNC_TO_LDS_B128
//            pipeline -> ds_load_b128 -> v_wmma_f32_16x16x32_bf16 (x3/K-step).
//   Stage 2: per-channel 7x7 window softmax-attention (VALU + v_exp_f32).
// Working set ~20MB (L2-resident); packing removes the 16x-redundant
// stride-4KB b32 gather of the B operand seen in the previous round.
// ---------------------------------------------------------------------------

typedef __attribute__((ext_vector_type(16))) __bf16 v16bf;
typedef __attribute__((ext_vector_type(8)))  float  v8f;

union OpU { uint4 q[2]; v16bf v; };   // one WMMA operand per lane (32B)

#define CC   256
#define HW   1024
#define NBAT 4

// lane-layout helpers (ISA 7.12.2, 16-bit A 16x32 / mirrored B 32x16):
//   row/col = lane&15 ; hk = (lane>>4)*8 ; elem i<8 -> K=hk+i ; i>=8 -> K=16+hk+(i-8)

// ---------------------------------------------------------------------------
// Stage 0a: pack wq/wk/wv -> apk[(m*16+mtile)*8+ks][lane] (32B per lane slot).
// ---------------------------------------------------------------------------
__global__ __launch_bounds__(256) void pack_w_kernel(
    const float* __restrict__ wq, const float* __restrict__ wk,
    const float* __restrict__ wv, uint4* __restrict__ apk)
{
    const int t    = blockIdx.x * 256 + threadIdx.x;   // 0..12287
    const int lane = t & 31;
    const int ks   = (t >> 5) & 7;
    const int mt   = (t >> 8) & 15;
    const int m    = t >> 12;                          // block-uniform
    const float* __restrict__ wp = (m == 0) ? wq : ((m == 1) ? wk : wv);

    const int hk  = (lane >> 4) << 3;
    const int row = mt * 16 + (lane & 15);
    const float* __restrict__ src = wp + row * CC + ks * 32;

    OpU o;
#pragma unroll
    for (int i = 0; i < 8; ++i) {
        o.v[i]     = (__bf16)src[hk + i];
        o.v[i + 8] = (__bf16)src[16 + hk + i];
    }
    apk[t * 2]     = o.q[0];
    apk[t * 2 + 1] = o.q[1];
}

// ---------------------------------------------------------------------------
// Stage 0b: pack x -> xpk[((b*64+nt)*8+ks)][lane] (32B per lane slot).
// ---------------------------------------------------------------------------
__global__ __launch_bounds__(256) void pack_x_kernel(
    const float* __restrict__ x, uint4* __restrict__ xpk)
{
    const int t    = blockIdx.x * 256 + threadIdx.x;   // 0..65535
    const int lane = t & 31;
    const int ks   = (t >> 5) & 7;
    const int nt   = (t >> 8) & 63;
    const int b    = t >> 14;

    const int hk  = (lane >> 4) << 3;
    const int col = nt * 16 + (lane & 15);
    const float* __restrict__ src = x + (b * CC + ks * 32) * HW + col;

    OpU o;
#pragma unroll
    for (int i = 0; i < 8; ++i) {
        o.v[i]     = (__bf16)src[(hk + i) * HW];
        o.v[i + 8] = (__bf16)src[(16 + hk + i) * HW];
    }
    xpk[t * 2]     = o.q[0];
    xpk[t * 2 + 1] = o.q[1];
}

// ---------------------------------------------------------------------------
// Async copy: one 16B chunk global -> LDS (CDNA5 async path, ASYNCcnt).
// ---------------------------------------------------------------------------
__device__ __forceinline__ void async_cp16(const uint4* lds_dst, const uint4* gsrc)
{
    unsigned           loff = (unsigned)(size_t)lds_dst;           // LDS byte addr
    unsigned long long gaddr = (unsigned long long)(size_t)gsrc;   // global VA
    asm volatile("global_load_async_to_lds_b128 %0, %1, off"
                 :: "v"(loff), "v"(gaddr) : "memory");
}

// ---------------------------------------------------------------------------
// Stage 1: q/k/v = {wq,wk,wv} @ x.  One wave per 16(M)x16(N) tile, K=256 in
// 8 steps of 32.  Double-buffered async-to-LDS pipeline:
//   issue(ks+1) -> s_wait_asynccnt 8 (retire ks batch, in-order) ->
//   ds_load_b128 operands -> 3x WMMA.
// ---------------------------------------------------------------------------
__global__ __launch_bounds__(256) void qkv_gemm_kernel(
    const uint4* __restrict__ apk,   // [3*16*8][32 lanes][2 x uint4]
    const uint4* __restrict__ xpk,   // [4*64*8][32 lanes][2 x uint4]
    float* __restrict__ qbuf, float* __restrict__ kbuf, float* __restrict__ vbuf)
{
    __shared__ uint4 lds[8 * 2 * 4 * 64];          // 8 waves x 2 bufs x 4 ops x 1KB = 64KB

    const int lane  = threadIdx.x & 31;
    const int wvid  = threadIdx.x >> 5;
    const int wid   = blockIdx.x * 8 + wvid;       // 0..4095
    const int mtile = wid & 15;
    const int ntile = wid >> 4;
    const int b     = ntile >> 6;
    const int nt    = ntile & 63;

    const int lbase = wvid * 512 + lane * 2;       // uint4 units

    v8f accq = {}; v8f acck = {}; v8f accv = {};

    auto issue = [&](int ks, int sel) {
        const uint4* gq = apk + (((0 * 16 + mtile) * 8) + ks) * 64 + lane * 2;
        const uint4* gk = apk + (((1 * 16 + mtile) * 8) + ks) * 64 + lane * 2;
        const uint4* gv = apk + (((2 * 16 + mtile) * 8) + ks) * 64 + lane * 2;
        const uint4* gx = xpk + (((b * 64 + nt) * 8) + ks) * 64 + lane * 2;
        uint4* l = (uint4*)lds + lbase + sel * 256;
        async_cp16(l + 0,   gq); async_cp16(l + 1,   gq + 1);
        async_cp16(l + 64,  gk); async_cp16(l + 65,  gk + 1);
        async_cp16(l + 128, gv); async_cp16(l + 129, gv + 1);
        async_cp16(l + 192, gx); async_cp16(l + 193, gx + 1);
    };

    issue(0, 0);
#pragma unroll
    for (int ks = 0; ks < 8; ++ks) {
        if (ks < 7) {
            issue(ks + 1, (ks + 1) & 1);
            asm volatile("s_wait_asynccnt 0x8" ::: "memory");  // ks batch retired
        } else {
            asm volatile("s_wait_asynccnt 0x0" ::: "memory");
        }
        const uint4* l = (const uint4*)lds + lbase + (ks & 1) * 256;
        OpU aq, ak, av, bx;
        aq.q[0] = l[0];   aq.q[1] = l[1];          // ds_load_b128 x8
        ak.q[0] = l[64];  ak.q[1] = l[65];
        av.q[0] = l[128]; av.q[1] = l[129];
        bx.q[0] = l[192]; bx.q[1] = l[193];

        accq = __builtin_amdgcn_wmma_f32_16x16x32_bf16(false, aq.v, false, bx.v,
                                                       (short)0, accq, false, false);
        acck = __builtin_amdgcn_wmma_f32_16x16x32_bf16(false, ak.v, false, bx.v,
                                                       (short)0, acck, false, false);
        accv = __builtin_amdgcn_wmma_f32_16x16x32_bf16(false, av.v, false, bx.v,
                                                       (short)0, accv, false, false);
    }

    // D layout: VGPR r -> M = r + 8*(lane>>4), N = lane&15.  Store channel-major.
    const int n     = lane & 15;
    const int mbase = mtile * 16 + ((lane >> 4) << 3);
    const int outb  = b * (CC * HW) + nt * 16 + n;
#pragma unroll
    for (int r = 0; r < 8; ++r) {
        const int idx = outb + (mbase + r) * HW;
        qbuf[idx] = accq[r];
        kbuf[idx] = acck[r];
        vbuf[idx] = accv[r];
    }
}

// ---------------------------------------------------------------------------
// Stage 2: per-(b,c,h,w) softmax over the 7x7 window.
//   thread = (w = tid&31, c wave-uniform); OOB: k=0 (bias still applied,
//   matching zero padding), v=0.
// ---------------------------------------------------------------------------
__global__ __launch_bounds__(256) void attn_window_kernel(
    const float* __restrict__ qbuf,
    const float* __restrict__ kbuf,
    const float* __restrict__ vbuf,
    const float* __restrict__ rel_h,   // [128][7]
    const float* __restrict__ rel_w,   // [128][7]
    float* __restrict__ out)           // [B][C][H][W]
{
    const int t   = threadIdx.x;
    const int w   = t & 31;
    const int cl  = t >> 5;
    const int blk = blockIdx.x;
    const int cg  = blk & 31;
    const int h   = (blk >> 5) & 31;
    const int b   = blk >> 10;
    const int c   = cg * 8 + cl;       // wave-uniform

    const bool isH = (c < 128);
    const float* __restrict__ rp = isH ? (rel_h + c * 7) : (rel_w + (c - 128) * 7);
    float rel[7];
#pragma unroll
    for (int i = 0; i < 7; ++i) rel[i] = rp[i];

    const int   base = (b * CC + c) * HW;
    const float qv   = qbuf[base + h * 32 + w];

    float s[49];
    float mx = -3.402823466e38f;
#pragma unroll
    for (int u = 0; u < 7; ++u) {
        const int  hh    = h + u - 3;
        const bool rowOK = (hh >= 0) & (hh < 32);
        const float bh   = rel[u];
#pragma unroll
        for (int v = 0; v < 7; ++v) {
            const int  ww = w + v - 3;
            const bool ok = rowOK & (ww >= 0) & (ww < 32);
            const float kv = ok ? kbuf[base + hh * 32 + ww] : 0.0f;
            const float sv = qv * (kv + (isH ? bh : rel[v]));
            s[u * 7 + v] = sv;
            mx = fmaxf(mx, sv);
        }
    }

    float sum = 0.0f;
#pragma unroll
    for (int j = 0; j < 49; ++j) {
        const float e = __expf(s[j] - mx);   // v_exp_f32
        s[j] = e;
        sum += e;
    }

    float acc = 0.0f;
#pragma unroll
    for (int u = 0; u < 7; ++u) {
        const int  hh    = h + u - 3;
        const bool rowOK = (hh >= 0) & (hh < 32);
#pragma unroll
        for (int v = 0; v < 7; ++v) {
            const int  ww = w + v - 3;
            const bool ok = rowOK & (ww >= 0) & (ww < 32);
            const float vv = ok ? vbuf[base + hh * 32 + ww] : 0.0f;
            acc += s[u * 7 + v] * vv;
        }
    }

    out[base + h * 32 + w] = acc / sum;
}

// ---------------------------------------------------------------------------
extern "C" void kernel_launch(void* const* d_in, const int* in_sizes, int n_in,
                              void* d_out, int out_size, void* d_ws, size_t ws_size,
                              hipStream_t stream) {
    const float* x     = (const float*)d_in[0];
    const float* wq    = (const float*)d_in[1];
    const float* wk    = (const float*)d_in[2];
    const float* wv    = (const float*)d_in[3];
    const float* rel_h = (const float*)d_in[4];
    const float* rel_w = (const float*)d_in[5];

    const size_t planeSz = (size_t)NBAT * CC * HW;     // 1M floats
    float* qbuf = (float*)d_ws;                         // 4MB
    float* kbuf = qbuf + planeSz;                       // 4MB
    float* vbuf = kbuf + planeSz;                       // 4MB
    uint4* apk  = (uint4*)(vbuf + planeSz);             // 3*16*8*64 uint4 = 384KB
    uint4* xpk  = apk + 3 * 16 * 8 * 64;                // 4*64*8*64 uint4 = 2MB

    pack_w_kernel<<<48,  256, 0, stream>>>(wq, wk, wv, apk);
    pack_x_kernel<<<256, 256, 0, stream>>>(x, xpk);
    // 4096 waves (16 M-tiles x 256 N-tiles), 8 waves / 256-thread block.
    qkv_gemm_kernel<<<512, 256, 0, stream>>>(apk, xpk, qbuf, kbuf, vbuf);
    attn_window_kernel<<<4096, 256, 0, stream>>>(qbuf, kbuf, vbuf, rel_h, rel_w,
                                                 (float*)d_out);
}
